// TextConditionedDynamicLayerAttention_82471962018677
// MI455X (gfx1250) — matrix operand order repaired and checked by reading, not compiled, not run on hardware
//
#include <hip/hip_runtime.h>
#include <hip/hip_bf16.h>
#include <math.h>

#define L 24
#define N 576
#define D 4096
#define T 128
#define RED 1024
#define FOCUS 22
#define EPS_LN 1e-5f
#define EPS_SC 1e-6f
#define THRESH 3.0f

// ---------------- WMMA types ----------------
typedef __attribute__((ext_vector_type(16))) __bf16 v16bf;
typedef __attribute__((ext_vector_type(8)))  float  v8f;

union Frag32B { uint4 u[2]; v16bf v; };

__device__ __forceinline__ unsigned short f2bf(float f) {
    union { float f; unsigned u; } v; v.f = f;
    unsigned r = v.u + 0x7FFFu + ((v.u >> 16) & 1u);
    return (unsigned short)(r >> 16);
}
__device__ __forceinline__ float sigmoidf_(float x) { return 1.0f / (1.0f + expf(-x)); }

__device__ __forceinline__ float blockReduceSum(float v, float* red) {
    int tid = threadIdx.x;
    red[tid] = v; __syncthreads();
    for (int s = blockDim.x >> 1; s > 0; s >>= 1) {
        if (tid < s) red[tid] += red[tid + s];
        __syncthreads();
    }
    float r = red[0]; __syncthreads();
    return r;
}
__device__ __forceinline__ float blockReduceMax(float v, float* red) {
    int tid = threadIdx.x;
    red[tid] = v; __syncthreads();
    for (int s = blockDim.x >> 1; s > 0; s >>= 1) {
        if (tid < s) red[tid] = fmaxf(red[tid], red[tid + s]);
        __syncthreads();
    }
    float r = red[0]; __syncthreads();
    return r;
}

// ---------------- stage 1: score_k_w fp32 -> bf16 (L2-resident B matrix) ----------------
__global__ void k_wcvt(const float* __restrict__ w, unsigned short* __restrict__ o) {
    size_t i = (size_t)blockIdx.x * blockDim.x + threadIdx.x;
    size_t stride = (size_t)gridDim.x * blockDim.x;
    for (size_t e = i; e < (size_t)D * D; e += stride) o[e] = f2bf(w[e]);
}

// ---------------- stage 2: per-(layer,group) sums (feeds y-mean AND 3x3 pooling) ----------------
__global__ void k_gsum(const float* __restrict__ lf, float* __restrict__ gsum) {
    int b = blockIdx.x;            // l*9 + g
    int l = b / 9, g = b % 9;
    int gi = g / 3, gk = g % 3;
    for (int d = threadIdx.x; d < D; d += blockDim.x) {
        float acc = 0.f;
        for (int j = 0; j < 8; ++j)
            for (int m = 0; m < 8; ++m) {
                int n = 192 * gi + 24 * j + 8 * gk + m;
                acc += lf[((size_t)(l * N + n)) * D + d];
            }
        gsum[(size_t)b * D + d] = acc;
    }
}

__global__ void k_pool(const float* __restrict__ gsum, float* __restrict__ y) {
    int idx = blockIdx.x * blockDim.x + threadIdx.x;   // l*D + d
    int l = idx >> 12, d = idx & (D - 1);
    float acc = 0.f;
    for (int g = 0; g < 9; ++g) acc += gsum[(size_t)(l * 9 + g) * D + d];
    y[(size_t)l * D + d] = acc * (1.0f / (float)N);
}

// ---------------- stage 3: text_global = LN(mean_t text_features) ----------------
__global__ void k_text(const float* __restrict__ tf, float* __restrict__ tg) {
    __shared__ float tmean[D];
    __shared__ float red[256];
    int tid = threadIdx.x;
    for (int d = tid; d < D; d += 256) {
        float acc = 0.f;
        for (int t = 0; t < T; ++t) acc += tf[(size_t)t * D + d];
        tmean[d] = acc * (1.0f / (float)T);
    }
    __syncthreads();
    float s = 0.f;
    for (int d = tid; d < D; d += 256) s += tmean[d];
    float mu = blockReduceSum(s, red) * (1.0f / (float)D);
    float s2 = 0.f;
    for (int d = tid; d < D; d += 256) { float dv = tmean[d] - mu; s2 += dv * dv; }
    float var = blockReduceSum(s2, red) * (1.0f / (float)D);
    float inv = 1.0f / sqrtf(var + EPS_LN);
    for (int d = tid; d < D; d += 256) tg[d] = (tmean[d] - mu) * inv;
}

// ---------------- stage 4: DSU recurrence (24 sequential steps) ----------------
__global__ void k_dsu_init(float* __restrict__ sigc, float* __restrict__ cbuf) {
    int d = blockIdx.x * blockDim.x + threadIdx.x;
    if (d < D) { sigc[d] = 0.5f; cbuf[d] = 0.0f; }
}

// s = relu(W1 @ [sig(c_prev); y_l; text_global] + b1), one row per block
__global__ void k_dsu_s(const float* __restrict__ W1, const float* __restrict__ b1,
                        const float* __restrict__ sigc, const float* __restrict__ y,
                        const float* __restrict__ tg, float* __restrict__ s, int l) {
    __shared__ float red[128];
    int row = blockIdx.x;
    const float* wr = W1 + (size_t)row * (3 * D);
    float acc = 0.f;
    for (int k = threadIdx.x; k < 3 * D; k += 128) {
        float v;
        if (k < D)            v = sigc[k];
        else if (k < 2 * D)   v = y[(size_t)l * D + (k - D)];
        else                  v = tg[k - 2 * D];
        acc += wr[k] * v;
    }
    float tot = blockReduceSum(acc, red);
    if (threadIdx.x == 0) s[row] = fmaxf(tot + b1[row], 0.0f);
}

// gates: c = f*c_prev + i*tanh(...), one output d per block
__global__ void k_dsu_gate(const float* __restrict__ Wc, const float* __restrict__ bclin, const float* __restrict__ bc,
                           const float* __restrict__ Wi, const float* __restrict__ bilin, const float* __restrict__ bi,
                           const float* __restrict__ Wf, const float* __restrict__ bflin, const float* __restrict__ bfv,
                           const float* __restrict__ s, float* __restrict__ cbuf,
                           float* __restrict__ sigc, float* __restrict__ ctx, int l) {
    __shared__ float red[128];
    int d = blockIdx.x;
    float ac = 0.f, ai = 0.f, af = 0.f;
    for (int k = threadIdx.x; k < RED; k += 128) {
        float sv = s[k];
        ac += Wc[(size_t)d * RED + k] * sv;
        ai += Wi[(size_t)d * RED + k] * sv;
        af += Wf[(size_t)d * RED + k] * sv;
    }
    float tc = blockReduceSum(ac, red);
    float ti = blockReduceSum(ai, red);
    float tf_ = blockReduceSum(af, red);
    if (threadIdx.x == 0) {
        float ct = tanhf(tc + bclin[d] + bc[d]);
        float iv = sigmoidf_(ti + bilin[d] + bi[d]);
        float fv = sigmoidf_(tf_ + bflin[d] + bfv[d]);
        float c  = fv * cbuf[d] + iv * ct;
        ctx[(size_t)l * D + d] = c;
        cbuf[d] = c;
        sigc[d] = sigmoidf_(c);
    }
}

// ---------------- stage 5: q_raw[l,d] = ctx[l] . score_q_w[d] (wave-per-output) ----------------
__global__ void k_q(const float* __restrict__ ctx, const float* __restrict__ qw,
                    float* __restrict__ qraw) {
    int gid = blockIdx.x * (blockDim.x >> 5) + (threadIdx.x >> 5); // wave index
    int lane = threadIdx.x & 31;
    int l = gid >> 12, d = gid & (D - 1);
    const float* cr = ctx + (size_t)l * D;
    const float* wr = qw + (size_t)d * D;
    float acc = 0.f;
    for (int k = lane; k < D; k += 32) acc += cr[k] * wr[k];
    for (int off = 16; off; off >>= 1) acc += __shfl_down(acc, off, 32);
    if (lane == 0) qraw[(size_t)l * D + d] = acc;
}

// LN(q_raw)*g+b folded into gq = g*qv, GQ = sum(g*qv), BQ = sum(b*qv)
__global__ void k_qln(const float* __restrict__ qraw, const float* __restrict__ g,
                      const float* __restrict__ b, float* __restrict__ gq,
                      float* __restrict__ GQ, float* __restrict__ BQ) {
    __shared__ float red[256];
    int l = blockIdx.x;
    const float* qr = qraw + (size_t)l * D;
    float s = 0.f, s2 = 0.f;
    for (int d = threadIdx.x; d < D; d += 256) { float v = qr[d]; s += v; s2 += v * v; }
    float mu = blockReduceSum(s, red) * (1.0f / (float)D);
    float var = blockReduceSum(s2, red) * (1.0f / (float)D) - mu * mu;
    float inv = 1.0f / sqrtf(var + EPS_LN);
    float gql = 0.f, bql = 0.f;
    for (int d = threadIdx.x; d < D; d += 256) {
        float qv = (qr[d] - mu) * inv * g[d] + b[d];
        float gv = g[d] * qv;
        gq[(size_t)l * D + d] = gv;
        gql += gv;
        bql += b[d] * qv;
    }
    float tg = blockReduceSum(gql, red);
    float tb = blockReduceSum(bql, red);
    if (threadIdx.x == 0) { GQ[l] = tg; BQ[l] = tb; }
}

// ---------------- stage 6: THE GEMM. bf16 WMMA, LN+score fused, kf never stored ----------------
// block = 256 (8 wave32), grid = 24 layers * 18 tiles of 32 rows.
// LDS: A tile 32x4096 bf16 (256KB) + gq (16KB) + 3x32 row accumulators.
#define BIG_SMEM (32 * D * 2 + D * 4 + 3 * 32 * 4)

__global__ void k_big(const float* __restrict__ lf, const unsigned short* __restrict__ wbf,
                      const float* __restrict__ gq, const float* __restrict__ GQ,
                      const float* __restrict__ BQ, float* __restrict__ raw) {
    extern __shared__ char smem[];
    unsigned short* Alds = (unsigned short*)smem;             // 32*4096 bf16
    float* gql = (float*)(smem + 32 * D * 2);                 // 4096 f32
    float* S1a = gql + D;                                     // 32
    float* S2a = S1a + 32;
    float* Za  = S2a + 32;

    const int tid  = threadIdx.x;
    const int wave = tid >> 5;
    const int lane = tid & 31;
    const int h    = wave & 1;      // row half: 0 -> rows 0..15, 1 -> rows 16..31
    const int dgrp = wave >> 1;     // d-stripe 0..3
    const int m    = lane & 15;
    const int sub  = lane >> 4;

    const int l  = blockIdx.x / 18;
    const int n0 = (blockIdx.x % 18) * 32;

    // stage A tile into LDS as bf16 (the single fp32 pass over layer_feats rows)
    const float4* Asrc = (const float4*)(lf + ((size_t)(l * N + n0)) * D);
    for (int i = tid; i < 32 * D / 4; i += 256) {
        float4 f = Asrc[i];
        unsigned long long p = (unsigned long long)f2bf(f.x)
                             | ((unsigned long long)f2bf(f.y) << 16)
                             | ((unsigned long long)f2bf(f.z) << 32)
                             | ((unsigned long long)f2bf(f.w) << 48);
        *(unsigned long long*)&Alds[i * 4] = p;
    }
    for (int i = tid; i < D; i += 256) gql[i] = gq[(size_t)l * D + i];
    if (tid < 96) S1a[tid] = 0.0f;   // zeroes S1a,S2a,Za (contiguous)
    __syncthreads();

    // per-lane running sums over all d for rows this lane owns (r in 0..7)
    float s1[8], s2[8], zz[8];
    #pragma unroll
    for (int r = 0; r < 8; ++r) { s1[r] = 0.f; s2[r] = 0.f; zz[r] = 0.f; }

    const uint4* ap = (const uint4*)(Alds + (size_t)(h * 16 + m) * D + sub * 8);

    for (int j = 0; j < 64; ++j) {
        const int dcol = dgrp * 16 + j * 64 + m;               // this lane's output column
        const uint4* bp = (const uint4*)(wbf + (size_t)dcol * D + sub * 8);
        v8f acc = {0.f, 0.f, 0.f, 0.f, 0.f, 0.f, 0.f, 0.f};
        for (int kk = 0; kk < 128; ++kk) {
            Frag32B a, b;
            a.u[0] = ap[kk * 4];         // K = k0+sub*8   .. +7
            a.u[1] = ap[kk * 4 + 2];     // K = k0+16+sub*8 .. +7
            b.u[0] = bp[kk * 4];         // K = k0+sub*8   .. +7
            b.u[1] = bp[kk * 4 + 2];     // K = k0+16+sub*8 .. +7
            __builtin_prefetch(bp + kk * 4 + 8, 0, 1);
            acc = __builtin_amdgcn_wmma_f32_16x16x32_bf16(
                false, a.v, false, b.v, (short)0, acc, false, false);
        }
        float gqs = gql[dcol];
        #pragma unroll
        for (int r = 0; r < 8; ++r) {
            float kv = acc[r];
            s1[r] += kv; s2[r] += kv * kv; zz[r] += kv * gqs;
        }
    }

    const int rbase = h * 16 + sub * 8;   // rows this lane contributed to
    #pragma unroll
    for (int r = 0; r < 8; ++r) {
        atomicAdd(&S1a[rbase + r], s1[r]);
        atomicAdd(&S2a[rbase + r], s2[r]);
        atomicAdd(&Za[rbase + r],  zz[r]);
    }
    __syncthreads();

    if (tid < 32) {
        float mu  = S1a[tid] * (1.0f / (float)D);
        float var = S2a[tid] * (1.0f / (float)D) - mu * mu;
        float inv = 1.0f / sqrtf(var + EPS_LN);
        raw[(size_t)l * N + n0 + tid] = (Za[tid] - mu * GQ[l]) * inv + BQ[l];
    }
}

// ---------------- stage 7: score normalization + conf ----------------
__global__ void k_scores(const float* __restrict__ raw, float* __restrict__ scores,
                         float* __restrict__ conf) {
    __shared__ float red[256];
    int l = blockIdx.x;
    const float* rr = raw + (size_t)l * N;
    float s = 0.f, s2 = 0.f;
    for (int n = threadIdx.x; n < N; n += 256) { float v = rr[n]; s += v; s2 += v * v; }
    float mu = blockReduceSum(s, red) * (1.0f / (float)N);
    float var = blockReduceSum(s2, red) * (1.0f / (float)N) - mu * mu;
    float sd = sqrtf(fmaxf(var, 0.0f)) + EPS_SC;
    float mx = -INFINITY;
    for (int n = threadIdx.x; n < N; n += 256) {
        float sc = (rr[n] - mu) / sd;
        scores[(size_t)l * N + n] = sc;
        mx = fmaxf(mx, sc);
    }
    float gmx = blockReduceMax(mx, red);
    if (threadIdx.x == 0) conf[l] = gmx;
}

// ---------------- stage 8: serial selection logic (L=24, trivial) ----------------
__global__ void k_sel(const float* __restrict__ conf, const float* __restrict__ logits,
                      int* __restrict__ li_o, int* __restrict__ r_o,
                      float* __restrict__ alpha) {
    if (threadIdx.x != 0 || blockIdx.x != 0) return;
    bool act_t[L]; int M_t = 0;
    float cn[L];
    for (int l = 0; l < L; ++l) {
        bool nf = (l != FOCUS);
        cn[l] = nf ? conf[l] : -INFINITY;
        act_t[l] = nf && (conf[l] > THRESH);
        if (act_t[l]) ++M_t;
    }
    // fallback top-8 (descending, earlier index on ties via strict >)
    int idx_f[8];
    float tmp[L]; for (int l = 0; l < L; ++l) tmp[l] = cn[l];
    for (int j = 0; j < 8; ++j) {
        float best = -INFINITY; int bi = 0;
        for (int l = 0; l < L; ++l) if (tmp[l] > best) { best = tmp[l]; bi = l; }
        idx_f[j] = bi; tmp[bi] = -INFINITY;
    }
    int pos_f[L], pos_t[L];
    bool act_f[L];
    for (int l = 0; l < L; ++l) { pos_f[l] = L; act_f[l] = false; }
    for (int j = 0; j < 8; ++j) { pos_f[idx_f[j]] = j; act_f[idx_f[j]] = true; }
    int c = 0;
    for (int l = 0; l < L; ++l) { if (act_t[l]) ++c; pos_t[l] = c - 1; }
    bool use_fb = (M_t < 8);
    int M = use_fb ? 8 : M_t;
    int base = 32 / M, rem = 32 - base * M;
    int k_nf[L], cum[L];
    int run = 0;
    for (int l = 0; l < L; ++l) {
        bool a = use_fb ? act_f[l] : act_t[l];
        int p = use_fb ? pos_f[l] : pos_t[l];
        k_nf[l] = a ? (base + ((p < rem) ? 1 : 0)) : 0;
        run += k_nf[l];
        cum[l] = run;
    }
    for (int t = 0; t < 32; ++t) {
        int li = 0;
        while (li < L && cum[li] <= t) ++li;
        li_o[t] = li;
        r_o[t] = t - (cum[li] - k_nf[li]);
    }
    // alpha = softmax(logits with FOCUS=-inf)
    float mx = -INFINITY;
    for (int l = 0; l < L; ++l) if (l != FOCUS) mx = fmaxf(mx, logits[l]);
    float sum = 0.f, e[L];
    for (int l = 0; l < L; ++l) {
        e[l] = (l == FOCUS) ? 0.0f : expf(logits[l] - mx);
        sum += e[l];
    }
    for (int l = 0; l < L; ++l) alpha[l] = e[l] / sum;
}

// ---------------- stage 9: per-layer iterative top-32 (desc, ties -> lower index) ----------------
__global__ void k_topk(const float* __restrict__ scores, int* __restrict__ topidx) {
    __shared__ float sv[N];
    int l = blockIdx.x, lane = threadIdx.x;   // 32 threads
    for (int n = lane; n < N; n += 32) sv[n] = scores[(size_t)l * N + n];
    __syncthreads();
    for (int it = 0; it < 32; ++it) {
        float bv = -INFINITY; int bi = 0;
        for (int n = lane; n < N; n += 32) {
            float v = sv[n];
            if (v > bv || (v == bv && n < bi)) { bv = v; bi = n; }
        }
        for (int off = 16; off; off >>= 1) {
            float ov = __shfl_down(bv, off, 32);
            int   oi = __shfl_down(bi, off, 32);
            if (ov > bv || (ov == bv && oi < bi)) { bv = ov; bi = oi; }
        }
        if (lane == 0) { topidx[l * 32 + it] = bi; sv[bi] = -INFINITY; }
        __syncthreads();
    }
}

// ---------------- stage 10: gather evidence rows ----------------
__global__ void k_gather(const float* __restrict__ lf, const int* __restrict__ topidx,
                         const int* __restrict__ li, const int* __restrict__ r,
                         float* __restrict__ out) {
    int j = blockIdx.x;
    int l, idx;
    if (j < 32) { l = FOCUS; idx = topidx[FOCUS * 32 + j]; }
    else        { int t = j - 32; l = li[t]; idx = topidx[l * 32 + r[t]]; }
    const float* src = lf + ((size_t)(l * N + idx)) * D;
    float* dst = out + (size_t)j * D;
    for (int k = threadIdx.x; k < D; k += 256) dst[k] = src[k];
}

// ---------------- stage 11: anchor-residual mix + group pooling from Gsum ----------------
__global__ void k_groups(const float* __restrict__ gsum, const float* __restrict__ alpha,
                         const float* __restrict__ scale, float* __restrict__ out) {
    int idx = blockIdx.x * blockDim.x + threadIdx.x;  // g*D + d
    int g = idx >> 12, d = idx & (D - 1);
    float mix = 0.f;
    for (int l = 0; l < L; ++l) mix += alpha[l] * gsum[(size_t)(l * 9 + g) * D + d];
    float anch = gsum[(size_t)(FOCUS * 9 + g) * D + d];
    float val = (anch + scale[0] * (mix - anch)) * (1.0f / 64.0f);
    out[(size_t)(64 + g) * D + d] = val;
}

// ---------------- host side ----------------
extern "C" void kernel_launch(void* const* d_in, const int* in_sizes, int n_in,
                              void* d_out, int out_size, void* d_ws, size_t ws_size,
                              hipStream_t stream) {
    const float* tf     = (const float*)d_in[0];
    const float* lf     = (const float*)d_in[1];
    const float* W1     = (const float*)d_in[2];
    const float* b1     = (const float*)d_in[3];
    const float* Wc     = (const float*)d_in[4];
    const float* bclin  = (const float*)d_in[5];
    const float* bc     = (const float*)d_in[6];
    const float* Wi     = (const float*)d_in[7];
    const float* bilin  = (const float*)d_in[8];
    const float* bi     = (const float*)d_in[9];
    const float* Wf     = (const float*)d_in[10];
    const float* bflin  = (const float*)d_in[11];
    const float* bfv    = (const float*)d_in[12];
    const float* qw     = (const float*)d_in[13];
    const float* kw     = (const float*)d_in[14];
    const float* lng    = (const float*)d_in[15];
    const float* lnb    = (const float*)d_in[16];
    const float* logits = (const float*)d_in[17];
    const float* scale  = (const float*)d_in[18];
    float* out = (float*)d_out;

    // workspace layout (floats)
    float* W = (float*)d_ws;
    size_t o = 0;
    float* gsum = W + o; o += (size_t)L * 9 * D;
    float* yb   = W + o; o += (size_t)L * D;
    float* tg   = W + o; o += D;
    float* sigc = W + o; o += D;
    float* cbuf = W + o; o += D;
    float* sbuf = W + o; o += RED;
    float* ctx  = W + o; o += (size_t)L * D;
    float* qraw = W + o; o += (size_t)L * D;
    float* gq   = W + o; o += (size_t)L * D;
    float* GQ   = W + o; o += 32;
    float* BQ   = W + o; o += 32;
    float* raw  = W + o; o += (size_t)L * N;
    float* scor = W + o; o += (size_t)L * N;
    float* conf = W + o; o += 32;
    float* alpha= W + o; o += 32;
    int* li_i   = (int*)(W + o); o += 32;
    int* r_i    = (int*)(W + o); o += 32;
    int* topidx = (int*)(W + o); o += L * 32;
    o = (o + 7) & ~(size_t)7;                    // 32B align for uint4 loads
    unsigned short* wbf = (unsigned short*)(W + o);  // D*D bf16 (32 MB)

    k_wcvt<<<8192, 256, 0, stream>>>(kw, wbf);
    k_gsum<<<L * 9, 256, 0, stream>>>(lf, gsum);
    k_pool<<<L * D / 256, 256, 0, stream>>>(gsum, yb);
    k_text<<<1, 256, 0, stream>>>(tf, tg);
    k_dsu_init<<<D / 256, 256, 0, stream>>>(sigc, cbuf);
    for (int l = 0; l < L; ++l) {
        k_dsu_s<<<RED, 128, 0, stream>>>(W1, b1, sigc, yb, tg, sbuf, l);
        k_dsu_gate<<<D, 128, 0, stream>>>(Wc, bclin, bc, Wi, bilin, bi,
                                          Wf, bflin, bfv, sbuf, cbuf, sigc, ctx, l);
    }
    k_q<<<L * D / 8, 256, 0, stream>>>(ctx, qw, qraw);
    k_qln<<<L, 256, 0, stream>>>(qraw, lng, lnb, gq, GQ, BQ);
    k_big<<<L * (N / 32), 256, BIG_SMEM, stream>>>(lf, wbf, gq, GQ, BQ, raw);
    k_scores<<<L, 256, 0, stream>>>(raw, scor, conf);
    k_sel<<<1, 1, 0, stream>>>(conf, logits, li_i, r_i, alpha);
    k_topk<<<L, 32, 0, stream>>>(scor, topidx);
    k_gather<<<64, 256, 0, stream>>>(lf, topidx, li_i, r_i, out);
    k_groups<<<9 * D / 256, 256, 0, stream>>>(gsum, alpha, scale, out);
}
